// EncoderStack_6528350290416
// MI455X (gfx1250) — matrix operand order repaired and checked by reading, not compiled, not run on hardware
//
#include <hip/hip_runtime.h>
#include <math.h>

// ---------------------------------------------------------------------------
// MoE-of-encoders (PatchTST-style) for MI455X / gfx1250.
// Heavy GEMMs run on v_wmma_f32_16x16x32_f16 (f16 in, f32 accumulate).
// All other math (gating DFT, softmax, LN, PE, exp/log) in f32/f64.
// ---------------------------------------------------------------------------

#define B_SZ 32
#define L_SEQ 96
#define D_MOD 32
#define N_HEADS 4
#define D_FF 128
#define N_EXP 18
#define EPS_F 2.220446049250313e-16f

typedef _Float16 v16h __attribute__((ext_vector_type(16)));
typedef _Float16 v8h  __attribute__((ext_vector_type(8)));
typedef _Float16 v4h  __attribute__((ext_vector_type(4)));
typedef float    v8f  __attribute__((ext_vector_type(8)));
typedef float    f4   __attribute__((ext_vector_type(4)));

// patch sizes ascending (np.unique order) -> expert index order of params
__device__ __constant__ int d_PS[N_EXP] = {2,3,4,5,6,7,8,9,10,11,12,14,16,20,24,32,48,96};
static const int h_PS[N_EXP] = {2,3,4,5,6,7,8,9,10,11,12,14,16,20,24,32,48,96};

// ---------------------------------------------------------------------------
// 1) Gating: amp = mean_d |rfft(x)| (freqs 1..48), group-sum -> softmax ->
//    top-3 renormalized gates. One block per batch element.
// ---------------------------------------------------------------------------
__global__ void gating_kernel(const float* __restrict__ x, float* __restrict__ gates) {
  int b = blockIdx.x;
  __shared__ float amp[48];
  __shared__ float logits[N_EXP];
  for (int i = threadIdx.x; i < 48; i += blockDim.x) amp[i] = 0.f;
  if (threadIdx.x < N_EXP) logits[threadIdx.x] = 0.f;
  __syncthreads();
  for (int idx = threadIdx.x; idx < 48 * 32; idx += blockDim.x) {
    int kk = idx >> 5;           // 0..47 -> freq k = kk+1
    int d  = idx & 31;
    float w = -2.0f * 3.14159265358979323846f * (float)(kk + 1) / 96.0f;
    float re = 0.f, im = 0.f;
    for (int t = 0; t < L_SEQ; ++t) {
      float val = x[((size_t)b * L_SEQ + t) * D_MOD + d];
      float sv, cv;
      sincosf(w * (float)t, &sv, &cv);
      re += val * cv;
      im += val * sv;
    }
    atomicAdd(&amp[kk], sqrtf(re * re + im * im) * (1.0f / 32.0f));
  }
  __syncthreads();
  if (threadIdx.x < 48) {
    int kk = threadIdx.x;
    int per = (L_SEQ + kk) / (kk + 1);      // ceil(96/(kk+1))
    int eidx = 0;
    for (int i = 0; i < N_EXP; ++i) if (d_PS[i] == per) eidx = i;
    atomicAdd(&logits[eidx], amp[kk]);
  }
  __syncthreads();
  if (threadIdx.x == 0) {
    float mx = -1e30f;
    for (int i = 0; i < N_EXP; ++i) mx = fmaxf(mx, logits[i]);
    float pr[N_EXP]; float sum = 0.f;
    for (int i = 0; i < N_EXP; ++i) { pr[i] = expf(logits[i] - mx); sum += pr[i]; }
    float inv = 1.0f / sum;
    for (int i = 0; i < N_EXP; ++i) pr[i] *= inv;
    int idxs[3]; float vals[3]; bool used[N_EXP] = {};
    for (int t = 0; t < 3; ++t) {
      int best = 0; float bv = -1e30f;
      for (int i = 0; i < N_EXP; ++i)
        if (!used[i] && pr[i] > bv) { bv = pr[i]; best = i; }
      used[best] = true; idxs[t] = best; vals[t] = bv;
    }
    float m2 = fmaxf(vals[0], fmaxf(vals[1], vals[2]));
    float tg[3]; float s2 = 0.f;
    for (int t = 0; t < 3; ++t) { tg[t] = expf(vals[t] - m2); s2 += tg[t]; }
    for (int i = 0; i < N_EXP; ++i) gates[b * N_EXP + i] = 0.f;
    for (int t = 0; t < 3; ++t) gates[b * N_EXP + idxs[t]] = tg[t] / s2;
  }
}

// ---------------------------------------------------------------------------
// 2) Positional embedding (sincos, f64, mean-0 / (std(ddof=1)*10)). One block.
// ---------------------------------------------------------------------------
__global__ void pe_kernel(float* __restrict__ pe, int Ntok, int dm) {
  __shared__ double red[256];
  __shared__ double red2[256];
  __shared__ double s_mean, s_scale;
  int total = Ntok * dm;
  double ls = 0.0, lq = 0.0;
  double lgc = log(10000.0) / (double)dm;
  for (int idx = threadIdx.x; idx < total; idx += 256) {
    int n = idx / dm, j = idx % dm;
    double dv = exp(-(double)(j & ~1) * lgc);
    double v = (j & 1) ? cos((double)n * dv) : sin((double)n * dv);
    ls += v; lq += v * v;
  }
  red[threadIdx.x] = ls; red2[threadIdx.x] = lq;
  __syncthreads();
  for (int s = 128; s > 0; s >>= 1) {
    if (threadIdx.x < s) { red[threadIdx.x] += red[threadIdx.x + s]; red2[threadIdx.x] += red2[threadIdx.x + s]; }
    __syncthreads();
  }
  if (threadIdx.x == 0) {
    double mean = red[0] / (double)total;
    double var  = (red2[0] - (double)total * mean * mean) / (double)(total - 1);
    s_mean = mean; s_scale = 1.0 / (sqrt(var) * 10.0);
  }
  __syncthreads();
  double mean = s_mean, scale = s_scale;
  for (int idx = threadIdx.x; idx < total; idx += 256) {
    int n = idx / dm, j = idx % dm;
    double dv = exp(-(double)(j & ~1) * lgc);
    double v = (j & 1) ? cos((double)n * dv) : sin((double)n * dv);
    pe[idx] = (float)((v - mean) * scale);
  }
}

// ---------------------------------------------------------------------------
// 3) Patchify with replicate padding: h[b, t, d] = x[b, min(t,95), d]
// ---------------------------------------------------------------------------
__global__ void patchify_kernel(const float* __restrict__ x, float* __restrict__ h,
                                int Npad, int total) {
  int idx = blockIdx.x * blockDim.x + threadIdx.x;
  if (idx >= total) return;
  int b = idx / (Npad * D_MOD);
  int r = idx % (Npad * D_MOD);
  int t = r >> 5, d = r & 31;
  int ts = t < L_SEQ ? t : (L_SEQ - 1);
  h[idx] = x[((size_t)b * L_SEQ + ts) * D_MOD + d];
}

// ---------------------------------------------------------------------------
// 4) WMMA GEMM: C[M,Nout] = act(A[M,K] @ W[K,Nout] + bias (+pe)).
//    Requirements (guaranteed by caller): K % 32 == 0, Nout % 32 == 0.
//    Block tile 128x32, 8 waves; each wave owns two 16x16 accumulators ->
//    two v_wmma_f32_16x16x32_f16 per K-step. f32 -> f16 conversion in LDS.
//    A rows are clamped (not masked): rows >= M feed only unstored outputs.
//    LDS layout: As[row][k] (8-half k-groups contiguous -> ds_load_b128),
//                Bs[col][k] transposed  (8-half k-groups contiguous per lane).
//    flags: 1 = add pe[(row % Ntok), col], 2 = relu
// ---------------------------------------------------------------------------
#define BM 128
#define BN 32
#define BK 32
#define LDK 40   // padded k-stride in halves (80B: 16B-aligned, bank-skewed)

__global__ __launch_bounds__(256)
void gemm_wmma(const float* __restrict__ A, const float* __restrict__ W,
               const float* __restrict__ bias, const float* __restrict__ pe,
               float* __restrict__ C, int M, int K, int Nout, int Ntok, int flags) {
  __shared__ _Float16 As[BM][LDK];
  __shared__ _Float16 Bs[BN][LDK];
  int bm0 = blockIdx.x * BM, bn0 = blockIdx.y * BN;
  int tid = threadIdx.x;
  int lane = tid & 31;
  int wave = tid >> 5;
  int wm = wave & 3;           // 4 row-groups of 32 rows
  int wn = wave >> 2;          // 2 col-groups of 16 cols
  int lm = lane & 15, lg = lane >> 4;

  // A-load mapping: 2 threads per row, 16 contiguous floats each
  int ar  = tid >> 1;              // 0..127
  int ac  = (tid & 1) * 16;        // 0 or 16
  int agr = bm0 + ar; if (agr >= M) agr = M - 1;   // branchless clamp
  const float* Arow = A + (size_t)agr * K + ac;
  // B-load mapping: each wave loads 4 k-rows, lane = column (coalesced)
  int bc  = lane;                  // 0..31 column
  int br0 = wave * 4;              // k-rows 4*wave .. +3

  v8f acc0 = {}, acc1 = {};
  for (int k0 = 0; k0 < K; k0 += BK) {
    if (k0 + BK < K && (tid & 63) == 0)
      __builtin_prefetch(&W[(size_t)(k0 + BK) * Nout + bn0], 0, 1);
    // ---- A tile: 128x32, vector loads + f16 convert + b128 LDS stores ----
    {
      f4 a0 = *(const f4*)(Arow + k0);
      f4 a1 = *(const f4*)(Arow + k0 + 4);
      f4 a2 = *(const f4*)(Arow + k0 + 8);
      f4 a3 = *(const f4*)(Arow + k0 + 12);
      v8h h0, h1;
      h0[0]=(_Float16)a0[0]; h0[1]=(_Float16)a0[1]; h0[2]=(_Float16)a0[2]; h0[3]=(_Float16)a0[3];
      h0[4]=(_Float16)a1[0]; h0[5]=(_Float16)a1[1]; h0[6]=(_Float16)a1[2]; h0[7]=(_Float16)a1[3];
      h1[0]=(_Float16)a2[0]; h1[1]=(_Float16)a2[1]; h1[2]=(_Float16)a2[2]; h1[3]=(_Float16)a2[3];
      h1[4]=(_Float16)a3[0]; h1[5]=(_Float16)a3[1]; h1[6]=(_Float16)a3[2]; h1[7]=(_Float16)a3[3];
      *(v8h*)&As[ar][ac]     = h0;
      *(v8h*)&As[ar][ac + 8] = h1;
    }
    // ---- B tile: 32x32 transposed into Bs[col][k] ----
    {
      const float* Wb = W + (size_t)(k0 + br0) * Nout + bn0 + bc;
      float w0 = Wb[0];
      float w1 = Wb[(size_t)Nout];
      float w2 = Wb[(size_t)2 * Nout];
      float w3 = Wb[(size_t)3 * Nout];
      v4h hb;
      hb[0]=(_Float16)w0; hb[1]=(_Float16)w1; hb[2]=(_Float16)w2; hb[3]=(_Float16)w3;
      *(v4h*)&Bs[bc][br0] = hb;
    }
    __syncthreads();
    // ---- fragments: contiguous b128 LDS loads (ISA 7.12.2 layouts) ----
    int row0 = wm * 32 + lm;
    v8h a00 = *(const v8h*)&As[row0][lg * 8];
    v8h a01 = *(const v8h*)&As[row0][16 + lg * 8];
    v8h a10 = *(const v8h*)&As[row0 + 16][lg * 8];
    v8h a11 = *(const v8h*)&As[row0 + 16][16 + lg * 8];
    v8h b0  = *(const v8h*)&Bs[wn * 16 + lm][lg * 8];
    v8h b1  = *(const v8h*)&Bs[wn * 16 + lm][16 + lg * 8];
    v16h af0 = __builtin_shufflevector(a00, a01, 0,1,2,3,4,5,6,7,8,9,10,11,12,13,14,15);
    v16h af1 = __builtin_shufflevector(a10, a11, 0,1,2,3,4,5,6,7,8,9,10,11,12,13,14,15);
    v16h bf  = __builtin_shufflevector(b0,  b1,  0,1,2,3,4,5,6,7,8,9,10,11,12,13,14,15);
    acc0 = __builtin_amdgcn_wmma_f32_16x16x32_f16(false, af0, false, bf,
                                                  (short)0, acc0, false, false);
    acc1 = __builtin_amdgcn_wmma_f32_16x16x32_f16(false, af1, false, bf,
                                                  (short)0, acc1, false, false);
    __syncthreads();
  }
  // ---- epilogue: C VGPR r -> rows (tile_m + r) lanes 0-15, (+8) lanes 16-31
  int col = bn0 + wn * 16 + lm;
  float bv = bias ? bias[col] : 0.0f;
  int tm = bm0 + wm * 32;
#pragma unroll
  for (int r = 0; r < 8; ++r) {
    int row = tm + r + (lg << 3);
    if (row < M) {
      float v = acc0[r] + bv;
      if (flags & 1) v += pe[(size_t)(row % Ntok) * Nout + col];
      if (flags & 2) v = v > 0.f ? v : 0.f;
      C[(size_t)row * Nout + col] = v;
    }
  }
#pragma unroll
  for (int r = 0; r < 8; ++r) {
    int row = tm + 16 + r + (lg << 3);
    if (row < M) {
      float v = acc1[r] + bv;
      if (flags & 1) v += pe[(size_t)(row % Ntok) * Nout + col];
      if (flags & 2) v = v > 0.f ? v : 0.f;
      C[(size_t)row * Nout + col] = v;
    }
  }
}

// ---------------------------------------------------------------------------
// 5) Fused attention per (b, head): scores (scale 1/sqrt(D//H)=1/sqrt(8)),
//    row softmax, context. N <= 49 so scores fit LDS.
// ---------------------------------------------------------------------------
__global__ void attention_kernel(const float* __restrict__ q, const float* __restrict__ k,
                                 const float* __restrict__ v, float* __restrict__ ctx,
                                 int Ntok, int dm, int dh) {
  int b = blockIdx.x >> 2;
  int hh = blockIdx.x & 3;
  __shared__ float s[49 * 49];
  const float scale = 0.35355339059327373f;  // 1/sqrt(8)
  int base = hh * dh;
  for (int idx = threadIdx.x; idx < Ntok * Ntok; idx += blockDim.x) {
    int n = idx / Ntok, m = idx % Ntok;
    const float* qp = q + ((size_t)(b * Ntok + n)) * dm + base;
    const float* kp = k + ((size_t)(b * Ntok + m)) * dm + base;
    float a = 0.f;
    for (int i = 0; i < dh; ++i) a += qp[i] * kp[i];
    s[idx] = a * scale;
  }
  __syncthreads();
  if ((int)threadIdx.x < Ntok) {
    int n = threadIdx.x;
    float mx = -1e30f;
    for (int m = 0; m < Ntok; ++m) mx = fmaxf(mx, s[n * Ntok + m]);
    float sum = 0.f;
    for (int m = 0; m < Ntok; ++m) { float e = expf(s[n * Ntok + m] - mx); s[n * Ntok + m] = e; sum += e; }
    float inv = 1.0f / sum;
    for (int m = 0; m < Ntok; ++m) s[n * Ntok + m] *= inv;
  }
  __syncthreads();
  for (int idx = threadIdx.x; idx < Ntok * dh; idx += blockDim.x) {
    int n = idx / dh, i = idx % dh;
    float a = 0.f;
    for (int m = 0; m < Ntok; ++m)
      a += s[n * Ntok + m] * v[((size_t)(b * Ntok + m)) * dm + base + i];
    ctx[((size_t)(b * Ntok + n)) * dm + base + i] = a;
  }
}

// ---------------------------------------------------------------------------
// 6) LayerNorm over D=32 (one wave32 per row): h = LN(h + delta) * g + b
// ---------------------------------------------------------------------------
__global__ void ln_kernel(float* __restrict__ h, const float* __restrict__ delta,
                          const float* __restrict__ g, const float* __restrict__ bta,
                          int rows) {
  int wid = (blockIdx.x * blockDim.x + threadIdx.x) >> 5;
  int lane = threadIdx.x & 31;
  if (wid >= rows) return;
  size_t off = (size_t)wid * 32 + lane;
  float x = h[off];
  if (delta) x += delta[off];
  float s = x;
#pragma unroll
  for (int o = 16; o > 0; o >>= 1) s += __shfl_xor(s, o, 32);
  float mu = s * (1.0f / 32.0f);
  float d = x - mu;
  float v2 = d * d;
#pragma unroll
  for (int o = 16; o > 0; o >>= 1) v2 += __shfl_xor(v2, o, 32);
  float var = v2 * (1.0f / 32.0f);
  h[off] = d * rsqrtf(var + 1e-5f) * g[lane] + bta[lane];
}

// ---------------------------------------------------------------------------
// 7) combine += gate_e * exp(out_e[:, :96, :]);  8) finalize; 9) zero
// ---------------------------------------------------------------------------
__global__ void combine_kernel(const float* __restrict__ h, const float* __restrict__ gates,
                               float* __restrict__ combined, int Npad, int e) {
  int idx = blockIdx.x * blockDim.x + threadIdx.x;
  if (idx >= B_SZ * L_SEQ * D_MOD) return;
  int b = idx / (L_SEQ * D_MOD);
  int r = idx % (L_SEQ * D_MOD);
  int t = r >> 5, d = r & 31;
  float gate = gates[b * N_EXP + e];
  combined[idx] += gate * expf(h[((size_t)b * Npad + t) * 32 + d]);
}

__global__ void final_kernel(const float* __restrict__ combined, float* __restrict__ out, int n) {
  int idx = blockIdx.x * blockDim.x + threadIdx.x;
  if (idx >= n) return;
  float c = combined[idx];
  out[idx] = logf(c == 0.0f ? EPS_F : c);
}

__global__ void zero_kernel(float* __restrict__ p, int n) {
  int idx = blockIdx.x * blockDim.x + threadIdx.x;
  if (idx < n) p[idx] = 0.f;
}

// ---------------------------------------------------------------------------
// Host orchestration. Param leaf order per layer dict:
// embed.w,b q.w,b k.w,b v.w,b o.w,b ln1.g,b fc1.w,b fc2.w,b ln2.g,b  (18)
// per expert: 2 layers (36) + norm.g,b (38). d_in[0] = x.
// ---------------------------------------------------------------------------
extern "C" void kernel_launch(void* const* d_in, const int* in_sizes, int n_in,
                              void* d_out, int out_size, void* d_ws, size_t ws_size,
                              hipStream_t stream) {
  (void)in_sizes; (void)n_in; (void)ws_size;
  const float* x = (const float*)d_in[0];
  float* ws = (float*)d_ws;

  const int ACT = B_SZ * 192 * 32;            // max B*Npad*32 = 196608
  float* gates    = ws;                        // 576
  float* combined = gates + 576;               // 98304
  float* pe       = combined + 98304;          // 6144 (max 2*3072)
  float* h   = pe + 6144;
  float* z   = h + ACT;
  float* qb  = z + ACT;
  float* kb  = qb + ACT;
  float* vb  = kb + ACT;
  float* ob  = vb + ACT;
  float* mid = ob + ACT;                       // B*Npad*128 <= 786432

  gating_kernel<<<B_SZ, 256, 0, stream>>>(x, gates);
  zero_kernel<<<(98304 + 255) / 256, 256, 0, stream>>>(combined, 98304);

  for (int e = 0; e < N_EXP; ++e) {
    int p    = h_PS[e];
    int N    = L_SEQ / p + 1;     // token count (full extra patch if divisible)
    int Npad = N * p;
    int dm   = p * D_MOD;
    int dh   = dm / N_HEADS;
    int M    = B_SZ * N;          // token rows
    int rows = B_SZ * Npad;       // per-timestep rows (LN / FFN)
    int htot = rows * D_MOD;

    pe_kernel<<<1, 256, 0, stream>>>(pe, N, dm);
    patchify_kernel<<<(htot + 255) / 256, 256, 0, stream>>>(x, h, Npad, htot);

    for (int l = 0; l < 2; ++l) {
      int base = 1 + e * 38 + l * 18;
      const float* Wemb = (const float*)d_in[base + 0];
      const float* bemb = (const float*)d_in[base + 1];
      const float* Wq   = (const float*)d_in[base + 2];
      const float* bq   = (const float*)d_in[base + 3];
      const float* Wk   = (const float*)d_in[base + 4];
      const float* bk   = (const float*)d_in[base + 5];
      const float* Wv   = (const float*)d_in[base + 6];
      const float* bv   = (const float*)d_in[base + 7];
      const float* Wo   = (const float*)d_in[base + 8];
      const float* bo   = (const float*)d_in[base + 9];
      const float* g1   = (const float*)d_in[base + 10];
      const float* b1   = (const float*)d_in[base + 11];
      const float* Wf1  = (const float*)d_in[base + 12];
      const float* bf1  = (const float*)d_in[base + 13];
      const float* Wf2  = (const float*)d_in[base + 14];
      const float* bf2  = (const float*)d_in[base + 15];
      const float* g2   = (const float*)d_in[base + 16];
      const float* b2   = (const float*)d_in[base + 17];

      dim3 gq((M + BM - 1) / BM, dm / BN);
      // z = h(M,dm) @ Wemb + b + pe
      gemm_wmma<<<gq, 256, 0, stream>>>(h, Wemb, bemb, pe, z, M, dm, dm, N, 1);
      gemm_wmma<<<gq, 256, 0, stream>>>(z, Wq, bq, nullptr, qb, M, dm, dm, N, 0);
      gemm_wmma<<<gq, 256, 0, stream>>>(z, Wk, bk, nullptr, kb, M, dm, dm, N, 0);
      gemm_wmma<<<gq, 256, 0, stream>>>(z, Wv, bv, nullptr, vb, M, dm, dm, N, 0);
      attention_kernel<<<B_SZ * N_HEADS, 256, 0, stream>>>(qb, kb, vb, z, N, dm, dh);
      gemm_wmma<<<gq, 256, 0, stream>>>(z, Wo, bo, nullptr, ob, M, dm, dm, N, 0);
      ln_kernel<<<(rows + 7) / 8, 256, 0, stream>>>(h, ob, g1, b1, rows);
      dim3 gf1((rows + BM - 1) / BM, D_FF / BN);
      gemm_wmma<<<gf1, 256, 0, stream>>>(h, Wf1, bf1, nullptr, mid, rows, D_MOD, D_FF, N, 2);
      dim3 gf2((rows + BM - 1) / BM, D_MOD / BN);
      gemm_wmma<<<gf2, 256, 0, stream>>>(mid, Wf2, bf2, nullptr, ob, rows, D_FF, D_MOD, N, 0);
      ln_kernel<<<(rows + 7) / 8, 256, 0, stream>>>(h, ob, g2, b2, rows);
    }
    const float* ng = (const float*)d_in[1 + e * 38 + 36];
    const float* nb = (const float*)d_in[1 + e * 38 + 37];
    ln_kernel<<<(rows + 7) / 8, 256, 0, stream>>>(h, nullptr, ng, nb, rows);
    combine_kernel<<<(98304 + 255) / 256, 256, 0, stream>>>(h, gates, combined, Npad, e);
  }
  final_kernel<<<(out_size + 255) / 256, 256, 0, stream>>>(combined, (float*)d_out, out_size);
}
